// DKNN_34875134443866
// MI455X (gfx1250) — matrix operand order repaired and checked by reading, not compiled, not run on hardware
//
#include <hip/hip_runtime.h>

// ---------------------------------------------------------------------------
// DKNN forward for gfx1250 (MI455X). All GEMMs via v_wmma_f32_16x16x32_bf16.
// GEMM uses 2x2 register blocking per wave (32x32/wave, 64x64/block) to double
// arithmetic intensity vs the naive 1-tile-per-wave version.
// ---------------------------------------------------------------------------

typedef __attribute__((ext_vector_type(16))) __bf16 v16bf;
typedef __attribute__((ext_vector_type(8)))  __bf16 v8bf;
typedef __attribute__((ext_vector_type(8)))  float  v8f;

union Frag16 { v16bf v; v8bf h[2]; };

#define BB   2048      // batch (unknown points)
#define NN   2048      // known points
#define DD   128       // d_model
#define KT   8         // d_trend
#define N2   2056      // NN + KT
#define LDA  4112      // 2*N2 (augmented)
#define TOPN 256
#define K2EXT 16512    // 128*128 + 128 (bias rows)

__device__ __forceinline__ unsigned short f2bf(float f) {
    unsigned int u = __float_as_uint(f);
    unsigned int r = u + 0x7FFFu + ((u >> 16) & 1u);
    return (unsigned short)(r >> 16);
}

// ---------------------------------------------------------------------------
// bf16 WMMA GEMM:  C[M,N] = scale * A[M,K] @ Bt[N,K]^T  (+ Cin)
// A,Bt row-major bf16 (Bt is B transposed). M,N multiples of 64, K of 32.
// Block = 128 threads = 4 waves (2x2); each wave computes a 32x32 tile as a
// 2x2 grid of 16x16 WMMA tiles -> 4 v_wmma per k-step on 4 fragments.
// Fragment layouts per CDNA5 ISA 7.12.2 (wave32).
// ---------------------------------------------------------------------------
__global__ void k_gemm_bf16(const unsigned short* __restrict__ A,
                            const unsigned short* __restrict__ Bt,
                            const float* __restrict__ Cin,
                            float* __restrict__ C,
                            int M, int N, int K, float scale) {
    int wave = threadIdx.x >> 5;
    int lane = threadIdx.x & 31;
    int tm = blockIdx.y * 64 + (wave >> 1) * 32;
    int tn = blockIdx.x * 64 + (wave & 1) * 32;
    int hi  = lane >> 4;     // 0 or 1 (lane half)
    int l15 = lane & 15;

    // A and B: K runs {base..base+7, base+16..base+23}, base = 8*half
    const unsigned short* ap0 = A  + (size_t)(tm + l15) * K + hi * 8;
    const unsigned short* ap1 = A  + (size_t)(tm + 16 + l15) * K + hi * 8;
    const unsigned short* bp0 = Bt + (size_t)(tn + l15) * K + hi * 8;
    const unsigned short* bp1 = Bt + (size_t)(tn + 16 + l15) * K + hi * 8;

    v8f acc00 = {0.f,0.f,0.f,0.f,0.f,0.f,0.f,0.f};
    v8f acc01 = acc00, acc10 = acc00, acc11 = acc00;

    for (int k = 0; k < K; k += 32) {
        Frag16 a0, a1, b0, b1;
        a0.h[0] = *reinterpret_cast<const v8bf*>(ap0 + k);
        a0.h[1] = *reinterpret_cast<const v8bf*>(ap0 + k + 16);
        a1.h[0] = *reinterpret_cast<const v8bf*>(ap1 + k);
        a1.h[1] = *reinterpret_cast<const v8bf*>(ap1 + k + 16);
        b0.h[0] = *reinterpret_cast<const v8bf*>(bp0 + k);
        b0.h[1] = *reinterpret_cast<const v8bf*>(bp0 + k + 16);
        b1.h[0] = *reinterpret_cast<const v8bf*>(bp1 + k);
        b1.h[1] = *reinterpret_cast<const v8bf*>(bp1 + k + 16);
        acc00 = __builtin_amdgcn_wmma_f32_16x16x32_bf16(false, a0.v, false, b0.v,
                                                        (short)0, acc00, false, false);
        acc01 = __builtin_amdgcn_wmma_f32_16x16x32_bf16(false, a0.v, false, b1.v,
                                                        (short)0, acc01, false, false);
        acc10 = __builtin_amdgcn_wmma_f32_16x16x32_bf16(false, a1.v, false, b0.v,
                                                        (short)0, acc10, false, false);
        acc11 = __builtin_amdgcn_wmma_f32_16x16x32_bf16(false, a1.v, false, b1.v,
                                                        (short)0, acc11, false, false);
    }

    v8f* accs[4] = { &acc00, &acc01, &acc10, &acc11 };
    #pragma unroll
    for (int tile = 0; tile < 4; ++tile) {
        int dm = (tile >> 1) * 16, dn = (tile & 1) * 16;
        size_t col = (size_t)tn + dn + l15;
        v8f& acc = *accs[tile];
        #pragma unroll
        for (int i = 0; i < 8; ++i) {
            size_t row = (size_t)tm + dm + i + hi * 8;
            size_t idx = row * (size_t)N + col;
            float v = scale * acc[i];
            if (Cin) v += Cin[idx];
            C[idx] = v;
        }
    }
}

// --------------------------- elementwise helpers ---------------------------

__global__ void k_f2bf(const float* x, unsigned short* y, int n) {
    int i = blockIdx.x * 256 + threadIdx.x;
    if (i < n) y[i] = f2bf(x[i]);
}

// Wt[dout,din] = bf16(W[din,dout]^T)
__global__ void k_w2bf(const float* W, unsigned short* Wt, int din, int dout) {
    int idx = blockIdx.x * 256 + threadIdx.x;
    if (idx >= din * dout) return;
    int o = idx / din, i = idx % din;
    Wt[idx] = f2bf(W[(size_t)i * dout + o]);
}

// x += bias (per col), optional PReLU, optional bf16 mirror
__global__ void k_bias_act(const float* x, const float* bias, float* xout,
                           unsigned short* bfout, int M, int N, int act) {
    int i = blockIdx.x * 256 + threadIdx.x;
    if (i >= M * N) return;
    float v = x[i] + bias[i % N];
    if (act) v = (v >= 0.f) ? v : 0.25f * v;
    xout[i] = v;
    if (bfout) bfout[i] = f2bf(v);
}

// position_rep
__global__ void k_pe(const float* coods, float* pe, int n) {
    int idx = blockIdx.x * 256 + threadIdx.x;
    if (idx >= n * 32) return;
    int b = idx >> 5, i = idx & 31;
    float theta = __expf(-4.f * (float)i / 128.f * 2.3025850929940457f) * 1.5707963267948966f;
    float ax = coods[b * 2 + 0] * theta;
    float ay = coods[b * 2 + 1] * theta;
    float sx, cx, sy, cy;
    __sincosf(ax, &sx, &cx);
    __sincosf(ay, &sy, &cy);
    size_t o = (size_t)b * 128 + 4 * i;
    pe[o + 0] = cx - sx;
    pe[o + 1] = sx + cx;
    pe[o + 2] = cy - sy;
    pe[o + 3] = sy + cy;
}

// q_in = 0.2*vec + 0.8*pe (f32 + bf16)
__global__ void k_mix(const float* vec, const float* pe, float* outf,
                      unsigned short* outbf, int n) {
    int i = blockIdx.x * 256 + threadIdx.x;
    if (i >= n) return;
    float v = vec[i] * 0.2f + pe[i] * 0.8f;
    outf[i] = v;
    outbf[i] = f2bf(v);
}

// per-row top-k threshold on sims (radix binary search in LDS) + masking
__global__ void k_topk_mask(const float* __restrict__ sims,
                            const float* __restrict__ scores,
                            float* __restrict__ outp,
                            int ncols, int topn, int zero_diag) {
    __shared__ unsigned int keys[2048];
    __shared__ int scnt;
    int row = blockIdx.x, t = threadIdx.x;
    for (int c = t; c < ncols; c += 256) {
        unsigned int b = __float_as_uint(sims[(size_t)row * ncols + c]);
        keys[c] = (b & 0x80000000u) ? ~b : (b | 0x80000000u);  // order-preserving
    }
    __syncthreads();
    unsigned int ans = 0u;
    for (int bit = 31; bit >= 0; --bit) {
        unsigned int cand = ans | (1u << bit);
        if (t == 0) scnt = 0;
        __syncthreads();
        int local = 0;
        for (int c = t; c < ncols; c += 256) local += (keys[c] >= cand);
        atomicAdd(&scnt, local);
        __syncthreads();
        if (scnt >= topn) ans = cand;
        __syncthreads();
    }
    for (int c = t; c < ncols; c += 256) {
        float s = scores[(size_t)row * ncols + c];
        bool keep = (keys[c] >= ans) && !(zero_diag && c == row);
        outp[(size_t)row * ncols + c] = keep ? s : 0.f;
    }
}

// GLU: h2 = g2 * sigmoid(g3) + res
__global__ void k_glu(const float* g2, const float* g3, const float* res,
                      float* h2, int n) {
    int i = blockIdx.x * 256 + threadIdx.x;
    if (i >= n) return;
    float s = 1.f / (1.f + __expf(-g3[i]));
    h2[i] = g2[i] * s + res[i];
}

// batchnorm stats over axis 0
__global__ void k_bn_stats(const float* h, float* mu, float* var, int M, int N) {
    __shared__ float s1[256], s2[256];
    int d = blockIdx.x, t = threadIdx.x;
    float a = 0.f, b = 0.f;
    for (int i = t; i < M; i += 256) {
        float v = h[(size_t)i * N + d];
        a += v; b += v * v;
    }
    s1[t] = a; s2[t] = b;
    __syncthreads();
    for (int s = 128; s > 0; s >>= 1) {
        if (t < s) { s1[t] += s1[t + s]; s2[t] += s2[t + s]; }
        __syncthreads();
    }
    if (t == 0) {
        float m = s1[0] / (float)M;
        mu[d] = m;
        var[d] = s2[0] / (float)M - m * m;
    }
}

// known_z[n] = mean_d( (h-mu)/sqrt(var+eps)*g + b )
__global__ void k_bn_rowmean(const float* h, const float* mu, const float* var,
                             const float* g, const float* bb, float* z, int N) {
    __shared__ float s1[128];
    int i = blockIdx.x, t = threadIdx.x;   // blockDim = 128 = N
    float v = (h[(size_t)i * N + t] - mu[t]) * rsqrtf(var[t] + 1e-5f) * g[t] + bb[t];
    s1[t] = v;
    __syncthreads();
    for (int s = 64; s > 0; s >>= 1) {
        if (t < s) s1[t] += s1[t + s];
        __syncthreads();
    }
    if (t == 0) z[i] = s1[0] / (float)N;
}

// metapn layer1: x1 = prelu(coods . w1 + l1b)
__global__ void k_meta_x1(const float* coods, const float* W1, const float* w1b,
                          const float* b1mat, const float* b1b, float* x1, int B) {
    int idx = blockIdx.x * 256 + threadIdx.x;
    if (idx >= B * 128) return;
    int b = idx >> 7, d = idx & 127;
    float wa = W1[(size_t)b * 256 + d]       + w1b[d];
    float wc = W1[(size_t)b * 256 + 128 + d] + w1b[128 + d];
    float v = coods[b * 2] * wa + coods[b * 2 + 1] * wc + b1mat[idx] + b1b[d];
    x1[idx] = (v >= 0.f) ? v : 0.25f * v;
}

// Mext[b, c*128+d] = pes[b,c]*x1[b,d];  Mext[b, 16384+d] = x1[b,d]
__global__ void k_buildM(const float* pes, const float* x1, unsigned short* Mx, int B) {
    int idx = blockIdx.x * 256 + threadIdx.x;
    if (idx >= B * K2EXT) return;
    int b = idx / K2EXT, k = idx % K2EXT;
    float v;
    if (k < 16384) v = pes[(size_t)b * 128 + (k >> 7)] * x1[(size_t)b * 128 + (k & 127)];
    else           v = x1[(size_t)b * 128 + (k - 16384)];
    Mx[(size_t)b * K2EXT + k] = f2bf(v);
}

// L2t[e, c*128+d] = l2w.w[c, d*128+e];  L2t[e, 16384+d] = l2w.b[d*128+e]
__global__ void k_repackL2(const float* W, const float* bias, unsigned short* L2t) {
    int idx = blockIdx.x * 256 + threadIdx.x;
    if (idx >= 128 * K2EXT) return;
    int e = idx / K2EXT, k = idx % K2EXT;
    float v;
    if (k < 16384) { int c = k >> 7, d = k & 127; v = W[(size_t)c * 16384 + d * 128 + e]; }
    else           { int d = k - 16384;           v = bias[(size_t)d * 128 + e]; }
    L2t[(size_t)e * K2EXT + k] = f2bf(v);
}

// x2 = prelu(X2 + b2mat + l2b.b)
__global__ void k_meta_x2(const float* X2, const float* b2mat, const float* b2b,
                          float* x2, unsigned short* x2bf, int B) {
    int idx = blockIdx.x * 256 + threadIdx.x;
    if (idx >= B * 128) return;
    float v = X2[idx] + b2mat[idx] + b2b[idx & 127];
    v = (v >= 0.f) ? v : 0.25f * v;
    x2[idx] = v;
    x2bf[idx] = f2bf(v);
}

// drift[b,t] = sum_d x2*(W3 + w3b) + pes@l3bw + l3bb
__global__ void k_meta_drift(const float* x2, const float* W3, const float* w3b,
                             const float* pes, const float* l3bw, const float* l3bb,
                             float* drift, int B) {
    int idx = blockIdx.x * 256 + threadIdx.x;
    if (idx >= B * KT) return;
    int b = idx >> 3, t = idx & 7;
    float acc = l3bb[t];
    for (int c = 0; c < 128; ++c) acc += pes[(size_t)b * 128 + c] * l3bw[c * KT + t];
    for (int d = 0; d < 128; ++d)
        acc += x2[(size_t)b * 128 + d] * (W3[(size_t)b * 1024 + d * KT + t] + w3b[d * KT + t]);
    drift[idx] = acc;
}

// assemble augmented kriging matrix [A | I], A = [[var_kn, drift],[drift^T,0]]
__global__ void k_buildAug(const float* varkn, const float* kdrift, float* Aug) {
    int idx = blockIdx.x * 256 + threadIdx.x;
    if (idx >= N2 * LDA) return;
    int i = idx / LDA, j = idx % LDA;
    float v;
    if (j < N2) {
        if (i < NN && j < NN)      v = varkn[(size_t)i * NN + j];
        else if (i < NN)           v = kdrift[(size_t)i * KT + (j - NN)];
        else if (j < NN)           v = kdrift[(size_t)j * KT + (i - NN)];
        else                       v = 0.f;
    } else {
        v = ((j - N2) == i) ? 1.f : 0.f;
    }
    Aug[(size_t)i * LDA + j] = v;
}

// Gauss-Jordan step k: partial pivot, row swap, scale, capture column k
__global__ void k_gj_pivot(float* Aug, float* colk, int k) {
    __shared__ float sval[256];
    __shared__ int   sidx[256];
    int t = threadIdx.x;
    float best = -1.f; int bi = k;
    for (int i = k + t; i < N2; i += 256) {
        float v = fabsf(Aug[(size_t)i * LDA + k]);
        if (v > best) { best = v; bi = i; }
    }
    sval[t] = best; sidx[t] = bi;
    __syncthreads();
    for (int s = 128; s > 0; s >>= 1) {
        if (t < s && sval[t + s] > sval[t]) { sval[t] = sval[t + s]; sidx[t] = sidx[t + s]; }
        __syncthreads();
    }
    int piv = sidx[0];
    if (piv != k) {
        for (int j = t; j < LDA; j += 256) {
            float a = Aug[(size_t)k * LDA + j];
            float b = Aug[(size_t)piv * LDA + j];
            Aug[(size_t)k * LDA + j] = b;
            Aug[(size_t)piv * LDA + j] = a;
        }
    }
    __syncthreads();
    float p = Aug[(size_t)k * LDA + k];
    float ip = (p != 0.f) ? 1.f / p : 0.f;
    for (int j = t; j < LDA; j += 256) Aug[(size_t)k * LDA + j] *= ip;
    __syncthreads();
    for (int i = t; i < N2; i += 256) colk[i] = (i == k) ? 0.f : Aug[(size_t)i * LDA + k];
}

__global__ void k_gj_elim(float* __restrict__ Aug, const float* __restrict__ colk, int k) {
    int i = blockIdx.y;
    int j = blockIdx.x * 256 + threadIdx.x;
    if (i == k || j >= LDA) return;
    float f = colk[i];
    if (f != 0.f) Aug[(size_t)i * LDA + j] -= f * Aug[(size_t)k * LDA + j];
}

// u[m] = sum_{j<NN} invA[j,m] * kz[j]   (invA = right half of Aug)
__global__ void k_uvec(const float* Aug, const float* kz, float* u) {
    int m = blockIdx.x * 256 + threadIdx.x;
    if (m >= N2) return;
    float acc = 0.f;
    for (int j = 0; j < NN; ++j)
        acc += Aug[(size_t)j * LDA + N2 + m] * kz[j];
    u[m] = acc;
}

// z_pre[b] = var_un[b,:] . u[:NN] + udrift[b,:] . u[NN:]
__global__ void k_zpre(const float* varun, const float* udrift, const float* u, float* z) {
    __shared__ float s1[256];
    int b = blockIdx.x, t = threadIdx.x;
    float a = 0.f;
    for (int j = t; j < NN; j += 256) a += varun[(size_t)b * NN + j] * u[j];
    if (t < KT) a += udrift[(size_t)b * KT + t] * u[NN + t];
    s1[t] = a;
    __syncthreads();
    for (int s = 128; s > 0; s >>= 1) {
        if (t < s) s1[t] += s1[t + s];
        __syncthreads();
    }
    if (t == 0) z[b] = s1[0];
}

__global__ void k_trend(const float* udrift, float* tout, int B) {
    int b = blockIdx.x * 256 + threadIdx.x;
    if (b >= B) return;
    float a = 0.f;
    for (int t = 0; t < KT; ++t) a += udrift[(size_t)b * KT + t];
    tout[b] = a / (float)KT;
}

// ---------------------------------------------------------------------------

extern "C" void kernel_launch(void* const* d_in, const int* in_sizes, int n_in,
                              void* d_out, int out_size, void* d_ws, size_t ws_size,
                              hipStream_t stream) {
    (void)in_sizes; (void)n_in; (void)out_size; (void)ws_size;
    // inputs (JAX pytree order: dict keys sorted alphabetically)
    const float* in_coods  = (const float*)d_in[0];
    const float* in_feat   = (const float*)d_in[1];
    const float* in_pe     = (const float*)d_in[2];
    const float* kn_coods  = (const float*)d_in[3];
    const float* kn_feat   = (const float*)d_in[4];
    const float* known_y   = (const float*)d_in[5];  (void)known_y; // unused in fwd
    const float* a_l1_b = (const float*)d_in[6],  *a_l1_w = (const float*)d_in[7];
    const float* a_l2_b = (const float*)d_in[8],  *a_l2_w = (const float*)d_in[9];
    const float* a_l3_b = (const float*)d_in[10], *a_l3_w = (const float*)d_in[11];
    const float* m_l1b_b = (const float*)d_in[12], *m_l1b_w = (const float*)d_in[13];
    const float* m_l1w_b = (const float*)d_in[14], *m_l1w_w = (const float*)d_in[15];
    const float* m_l2b_b = (const float*)d_in[16], *m_l2b_w = (const float*)d_in[17];
    const float* m_l2w_b = (const float*)d_in[18], *m_l2w_w = (const float*)d_in[19];
    const float* m_l3b_b = (const float*)d_in[20], *m_l3b_w = (const float*)d_in[21];
    const float* m_l3w_b = (const float*)d_in[22], *m_l3w_w = (const float*)d_in[23];
    const float* bn_b = (const float*)d_in[24], *bn_g = (const float*)d_in[25];
    const float* p_f1_b  = (const float*)d_in[26], *p_f1_w  = (const float*)d_in[27];
    const float* p_f12_b = (const float*)d_in[28], *p_f12_w = (const float*)d_in[29];
    const float* p_f2_b  = (const float*)d_in[30], *p_f2_w  = (const float*)d_in[31];
    const float* p_f3_b  = (const float*)d_in[32], *p_f3_w  = (const float*)d_in[33];
    const float* ssan_wk = (const float*)d_in[34], *ssan_wq = (const float*)d_in[35];

    // workspace bump allocator
    char* wp = (char*)d_ws;
    auto allocF = [&](size_t n) -> float* {
        float* p = (float*)wp; wp += ((n * 4 + 255) & ~(size_t)255); return p;
    };
    auto allocH = [&](size_t n) -> unsigned short* {
        unsigned short* p = (unsigned short*)wp; wp += ((n * 2 + 255) & ~(size_t)255); return p;
    };

    // f32 buffers
    float* pe_know = allocF((size_t)NN * DD);
    float* vec_un  = allocF((size_t)BB * DD);
    float* vec_kn  = allocF((size_t)NN * DD);
    float* t1      = allocF((size_t)BB * DD);
    float* t2      = allocF((size_t)BB * DD);
    float* qin_un  = allocF((size_t)BB * DD);
    float* kin_kn  = allocF((size_t)NN * DD);
    float* query_un = allocF((size_t)BB * DD);
    float* key_kn   = allocF((size_t)NN * DD);
    float* query_kn = allocF((size_t)NN * DD);
    float* scoresA  = allocF((size_t)BB * NN);   // scores_un -> var_un (in place)
    float* simsBuf  = allocF((size_t)BB * NN);
    float* scoresB  = allocF((size_t)NN * NN);   // scores_kn
    float* hbuf = allocF((size_t)NN * DD);
    float* g2   = allocF((size_t)NN * DD);
    float* g3   = allocF((size_t)NN * DD);
    float* h2   = allocF((size_t)NN * DD);
    float* mu   = allocF(DD);
    float* var  = allocF(DD);
    float* knz  = allocF(NN);
    float* W1    = allocF((size_t)BB * 256);
    float* b1mat = allocF((size_t)BB * DD);
    float* x1    = allocF((size_t)BB * DD);
    float* X2    = allocF((size_t)BB * DD);
    float* b2mat = allocF((size_t)BB * DD);
    float* x2    = allocF((size_t)BB * DD);
    float* W3    = allocF((size_t)BB * 1024);
    float* drift_kn = allocF((size_t)NN * KT);
    float* drift_un = allocF((size_t)BB * KT);
    float* Aug  = allocF((size_t)N2 * LDA);
    float* colk = allocF(N2);
    float* uvec = allocF(N2);

    // bf16 buffers
    unsigned short* w1t = allocH(128 * 32);
    unsigned short* w2t = allocH(128 * 128);
    unsigned short* w3t = allocH(128 * 128);
    unsigned short* wq_t = allocH(128 * 128);
    unsigned short* wk_t = allocH(128 * 128);
    unsigned short* f1t  = allocH(128 * 128);
    unsigned short* f12t = allocH(128 * 128);
    unsigned short* f2t  = allocH(128 * 128);
    unsigned short* f3t  = allocH(128 * 128);
    unsigned short* l1w_t = allocH(256 * 128);
    unsigned short* l1b_t = allocH(128 * 128);
    unsigned short* l2b_t = allocH(128 * 128);
    unsigned short* l3w_t = allocH(1024 * 128);
    unsigned short* L2t   = allocH((size_t)128 * K2EXT);
    unsigned short* feat_un_bf = allocH((size_t)BB * 32);
    unsigned short* feat_kn_bf = allocH((size_t)NN * 32);
    unsigned short* pe_un_bf = allocH((size_t)BB * DD);
    unsigned short* pe_kn_bf = allocH((size_t)NN * DD);
    unsigned short* vec_kn_bf = allocH((size_t)NN * DD);
    unsigned short* t1bf = allocH((size_t)BB * DD);
    unsigned short* t2bf = allocH((size_t)BB * DD);
    unsigned short* qin_un_bf = allocH((size_t)BB * DD);
    unsigned short* kin_kn_bf = allocH((size_t)NN * DD);
    unsigned short* query_un_bf = allocH((size_t)BB * DD);
    unsigned short* key_kn_bf   = allocH((size_t)NN * DD);
    unsigned short* query_kn_bf = allocH((size_t)NN * DD);
    unsigned short* h_bf  = allocH((size_t)NN * DD);
    unsigned short* x2bf  = allocH((size_t)BB * DD);
    unsigned short* Mext  = allocH((size_t)BB * K2EXT);

    float* out_z     = (float*)d_out;
    float* out_varkn = (float*)d_out + NN;
    float* out_trend = (float*)d_out + NN + (size_t)NN * NN;

    auto g1 = [](int n) { return dim3((n + 255) / 256); };
    auto gemm = [&](const unsigned short* A, const unsigned short* Bt, const float* Cin,
                    float* C, int M, int N, int K, float scale) {
        k_gemm_bf16<<<dim3(N / 64, M / 64), 128, 0, stream>>>(A, Bt, Cin, C, M, N, K, scale);
    };

    // ---- weight repacks (bf16, transposed) ----
    k_w2bf<<<g1(32 * 128), 256, 0, stream>>>(a_l1_w, w1t, 32, 128);
    k_w2bf<<<g1(128 * 128), 256, 0, stream>>>(a_l2_w, w2t, 128, 128);
    k_w2bf<<<g1(128 * 128), 256, 0, stream>>>(a_l3_w, w3t, 128, 128);
    k_w2bf<<<g1(128 * 128), 256, 0, stream>>>(ssan_wq, wq_t, 128, 128);
    k_w2bf<<<g1(128 * 128), 256, 0, stream>>>(ssan_wk, wk_t, 128, 128);
    k_w2bf<<<g1(128 * 128), 256, 0, stream>>>(p_f1_w,  f1t,  128, 128);
    k_w2bf<<<g1(128 * 128), 256, 0, stream>>>(p_f12_w, f12t, 128, 128);
    k_w2bf<<<g1(128 * 128), 256, 0, stream>>>(p_f2_w,  f2t,  128, 128);
    k_w2bf<<<g1(128 * 128), 256, 0, stream>>>(p_f3_w,  f3t,  128, 128);
    k_w2bf<<<g1(128 * 256), 256, 0, stream>>>(m_l1w_w, l1w_t, 128, 256);
    k_w2bf<<<g1(128 * 128), 256, 0, stream>>>(m_l1b_w, l1b_t, 128, 128);
    k_w2bf<<<g1(128 * 128), 256, 0, stream>>>(m_l2b_w, l2b_t, 128, 128);
    k_w2bf<<<g1(128 * 1024), 256, 0, stream>>>(m_l3w_w, l3w_t, 128, 1024);
    k_repackL2<<<g1(128 * K2EXT), 256, 0, stream>>>(m_l2w_w, m_l2w_b, L2t);

    // ---- position encoding + input conversions ----
    k_pe<<<g1(NN * 32), 256, 0, stream>>>(kn_coods, pe_know, NN);
    k_f2bf<<<g1(NN * DD), 256, 0, stream>>>(pe_know, pe_kn_bf, NN * DD);
    k_f2bf<<<g1(BB * DD), 256, 0, stream>>>(in_pe, pe_un_bf, BB * DD);
    k_f2bf<<<g1(BB * 32), 256, 0, stream>>>(in_feat, feat_un_bf, BB * 32);
    k_f2bf<<<g1(NN * 32), 256, 0, stream>>>(kn_feat, feat_kn_bf, NN * 32);

    // ---- attre (shared MLP), run for un and kn ----
    auto attre = [&](const unsigned short* fbf, float* vec, unsigned short* vbf, int M) {
        gemm(fbf, w1t, nullptr, t1, M, 128, 32, 1.f);
        k_bias_act<<<g1(M * 128), 256, 0, stream>>>(t1, a_l1_b, t1, t1bf, M, 128, 1);
        gemm(t1bf, w2t, nullptr, t2, M, 128, 128, 1.f);
        k_bias_act<<<g1(M * 128), 256, 0, stream>>>(t2, a_l2_b, t2, t2bf, M, 128, 1);
        gemm(t2bf, w3t, nullptr, vec, M, 128, 128, 1.f);
        k_bias_act<<<g1(M * 128), 256, 0, stream>>>(vec, a_l3_b, vec, vbf, M, 128, 0);
    };
    attre(feat_un_bf, vec_un, nullptr, BB);
    attre(feat_kn_bf, vec_kn, vec_kn_bf, NN);

    // ---- SSAN ----
    const float s128 = 0.08838834764831845f;  // 1/sqrt(128)
    k_mix<<<g1(BB * DD), 256, 0, stream>>>(vec_un, in_pe, qin_un, qin_un_bf, BB * DD);
    k_mix<<<g1(NN * DD), 256, 0, stream>>>(vec_kn, pe_know, kin_kn, kin_kn_bf, NN * DD);
    gemm(qin_un_bf, wq_t, qin_un, query_un, BB, 128, 128, 1.f);
    k_f2bf<<<g1(BB * DD), 256, 0, stream>>>(query_un, query_un_bf, BB * DD);
    gemm(kin_kn_bf, wk_t, kin_kn, key_kn, NN, 128, 128, 1.f);
    k_f2bf<<<g1(NN * DD), 256, 0, stream>>>(key_kn, key_kn_bf, NN * DD);
    gemm(kin_kn_bf, wq_t, kin_kn, query_kn, NN, 128, 128, 1.f);
    k_f2bf<<<g1(NN * DD), 256, 0, stream>>>(query_kn, query_kn_bf, NN * DD);

    gemm(query_un_bf, key_kn_bf, nullptr, scoresA, BB, NN, 128, s128);
    gemm(pe_un_bf, pe_kn_bf, nullptr, simsBuf, BB, NN, 128, s128);
    k_topk_mask<<<BB, 256, 0, stream>>>(simsBuf, scoresA, scoresA, NN, TOPN, 0);  // var_un

    gemm(query_kn_bf, key_kn_bf, nullptr, scoresB, NN, NN, 128, s128);
    gemm(pe_kn_bf, pe_kn_bf, nullptr, simsBuf, NN, NN, 128, s128);
    k_topk_mask<<<NN, 256, 0, stream>>>(simsBuf, scoresB, out_varkn, NN, TOPN, 1); // var_kn -> d_out

    // ---- PGRN -> known_z ----
    gemm(vec_kn_bf, f1t, nullptr, hbuf, NN, 128, 128, 1.f);
    gemm(pe_kn_bf, f12t, hbuf, hbuf, NN, 128, 128, 1.f);
    k_bias_act<<<g1(NN * DD), 256, 0, stream>>>(hbuf, p_f1_b, hbuf, nullptr, NN, 128, 0);
    k_bias_act<<<g1(NN * DD), 256, 0, stream>>>(hbuf, p_f12_b, hbuf, h_bf, NN, 128, 0);
    gemm(h_bf, f2t, nullptr, g2, NN, 128, 128, 1.f);
    k_bias_act<<<g1(NN * DD), 256, 0, stream>>>(g2, p_f2_b, g2, nullptr, NN, 128, 0);
    gemm(h_bf, f3t, nullptr, g3, NN, 128, 128, 1.f);
    k_bias_act<<<g1(NN * DD), 256, 0, stream>>>(g3, p_f3_b, g3, nullptr, NN, 128, 0);
    k_glu<<<g1(NN * DD), 256, 0, stream>>>(g2, g3, vec_kn, h2, NN * DD);
    k_bn_stats<<<DD, 256, 0, stream>>>(h2, mu, var, NN, DD);
    k_bn_rowmean<<<NN, 128, 0, stream>>>(h2, mu, var, bn_g, bn_b, knz, DD);

    // ---- metapn (hypernetwork as outer-product GEMM), run for kn and un ----
    auto metapn = [&](const float* pes, const unsigned short* pes_bf,
                      const float* coods, float* drift) {
        gemm(pes_bf, l1w_t, nullptr, W1, BB, 256, 128, 1.f);
        gemm(pes_bf, l1b_t, nullptr, b1mat, BB, 128, 128, 1.f);
        k_meta_x1<<<g1(BB * 128), 256, 0, stream>>>(coods, W1, m_l1w_b, b1mat, m_l1b_b, x1, BB);
        k_buildM<<<g1(BB * K2EXT), 256, 0, stream>>>(pes, x1, Mext, BB);
        gemm(Mext, L2t, nullptr, X2, BB, 128, K2EXT, 1.f);
        gemm(pes_bf, l2b_t, nullptr, b2mat, BB, 128, 128, 1.f);
        k_meta_x2<<<g1(BB * 128), 256, 0, stream>>>(X2, b2mat, m_l2b_b, x2, x2bf, BB);
        gemm(pes_bf, l3w_t, nullptr, W3, BB, 1024, 128, 1.f);
        k_meta_drift<<<g1(BB * KT), 256, 0, stream>>>(x2, W3, m_l3w_b, pes, m_l3b_w, m_l3b_b, drift, BB);
    };
    metapn(pe_know, pe_kn_bf, kn_coods, drift_kn);
    metapn(in_pe,   pe_un_bf, in_coods, drift_un);

    // ---- kriging: build A, invert via Gauss-Jordan (partial pivoting) ----
    k_buildAug<<<g1(N2 * LDA), 256, 0, stream>>>(out_varkn, drift_kn, Aug);
    for (int k = 0; k < N2; ++k) {
        k_gj_pivot<<<1, 256, 0, stream>>>(Aug, colk, k);
        k_gj_elim<<<dim3((LDA + 255) / 256, N2), 256, 0, stream>>>(Aug, colk, k);
    }

    // ---- z_pre = sys_pre @ inv(A).T restricted to known_z: two matvecs ----
    k_uvec<<<g1(N2), 256, 0, stream>>>(Aug, knz, uvec);
    k_zpre<<<BB, 256, 0, stream>>>(scoresA /*var_un*/, drift_un, uvec, out_z);
    k_trend<<<g1(BB), 256, 0, stream>>>(drift_un, out_trend, BB);
}
